// MegNetBlock_v2_55851754717349
// MI455X (gfx1250) — compile-verified, matrix operand order checked by reading, not run on hardware
//
#include <hip/hip_runtime.h>
#include <hip/hip_bf16.h>

// ---------------------------------------------------------------------------
// MegNet-style GNN forward for MI455X (gfx1250, wave32, WMMA).
//
//  * bf16 activations (BW-bound workload @ 23.3 TB/s), fp32 accumulation in
//    v_wmma_f32_16x16x32_bf16.
//  * Gather-concat GEMM templated on <KP, RELU>: A fragments register-cached
//    per wave (KP/32 x v16bf), B streamed from LDS; branch-free epilogue
//    (all GEMM outputs padded to 128-row multiples in workspace so stores
//    are unconditional; padding rows never consumed).
//  * BM=128 rows/WG, 256 threads = 8 waves. LDS = (128+N)*KP*2 <= 128KB.
//  * scatter_mean via fp32 atomics + count, then divide.
//
// Param flattening assumption: setup_inputs() dict insertion order (95 inputs).
// ---------------------------------------------------------------------------

typedef __attribute__((ext_vector_type(16))) __bf16 v16bf;
typedef __attribute__((ext_vector_type(8)))  __bf16 v8bf;
typedef __attribute__((ext_vector_type(8)))  float  v8f;

__device__ __forceinline__ __bf16 f2bf(float f) {
    unsigned u = __builtin_bit_cast(unsigned, f);
    unsigned r = u + 0x7FFFu + ((u >> 16) & 1u);   // round-to-nearest-even
    unsigned short h = (unsigned short)(r >> 16);
    return __builtin_bit_cast(__bf16, h);
}
__device__ __forceinline__ float bf2f(__bf16 h) {
    unsigned short s = __builtin_bit_cast(unsigned short, h);
    unsigned u = ((unsigned)s) << 16;
    return __builtin_bit_cast(float, u);
}
__device__ __forceinline__ __bf16 bfzero() {
    return __builtin_bit_cast(__bf16, (unsigned short)0);
}

// ---------------------------------------------------------------------------
// Small utility kernels
// ---------------------------------------------------------------------------
__global__ void k_pack_wT(__bf16* __restrict__ dst, const float* __restrict__ W,
                          int K, int N, int Kp) {
    int i = blockIdx.x * blockDim.x + threadIdx.x;
    if (i >= N * Kp) return;
    int n = i / Kp, k = i % Kp;
    dst[i] = (k < K) ? f2bf(W[(size_t)k * N + n]) : bfzero();
}
__global__ void k_f32_to_bf16(__bf16* __restrict__ dst, const float* __restrict__ src, int n) {
    int i = blockIdx.x * blockDim.x + threadIdx.x;
    if (i < n) dst[i] = f2bf(src[i]);
}
__global__ void k_bf16_to_f32(float* __restrict__ dst, const __bf16* __restrict__ src, int n) {
    int i = blockIdx.x * blockDim.x + threadIdx.x;
    if (i < n) dst[i] = bf2f(src[i]);
}
__global__ void k_zero_f32(float* __restrict__ p, int n) {
    int i = blockIdx.x * blockDim.x + threadIdx.x;
    if (i < n) p[i] = 0.f;
}
__global__ void k_batch_e(int* __restrict__ be, const int* __restrict__ row,
                          const int* __restrict__ batch, int E) {
    int i = blockIdx.x * blockDim.x + threadIdx.x;
    if (i < E) be[i] = batch[row[i]];
}
__global__ void k_resadd(__bf16* __restrict__ a, const __bf16* __restrict__ b, int n) {
    int i = blockIdx.x * blockDim.x + threadIdx.x;
    if (i < n) a[i] = f2bf(bf2f(a[i]) + bf2f(b[i]));
}
__global__ void k_scatter_feat(const __bf16* __restrict__ src, const int* __restrict__ idx,
                               float* __restrict__ acc, int rows) {
    int t = blockIdx.x * blockDim.x + threadIdx.x;
    if (t >= rows * 64) return;
    int r = t >> 6, c = t & 63;
    atomicAdd(&acc[(size_t)idx[r] * 64 + c], bf2f(src[t]));
}
__global__ void k_scatter_cnt(const int* __restrict__ idx, float* __restrict__ cnt, int rows) {
    int t = blockIdx.x * blockDim.x + threadIdx.x;
    if (t < rows) atomicAdd(&cnt[idx[t]], 1.f);
}
__global__ void k_mean_bf16(const float* __restrict__ acc, const float* __restrict__ cnt,
                            __bf16* __restrict__ out, int R) {
    int t = blockIdx.x * blockDim.x + threadIdx.x;
    if (t >= R * 64) return;
    float c = cnt[t >> 6];
    out[t] = f2bf(acc[t] / (c > 1.f ? c : 1.f));
}

// ---------------------------------------------------------------------------
// Fused gather-concat GEMM:  OUT[M x N] = act( A @ W + b )
// A row r = concat(seg0..seg3), seg.src bf16 [rows x width] gathered via
// seg.idx (nullptr => identity).  KP compile-time (mult of 32), N in {64,128}.
//
// LDS: As[128][KP] | Ws[N][KP] (W^T image, pre-packed bf16).
// Wave w handles A rows w*16..w*16+15:
//   - loads its KP/32 A-fragments from LDS into registers once,
//   - sweeps N/16 column tiles streaming only B fragments from LDS.
// Fragment packing per ISA 7.12.2 (lanes 0-15: K 0-7 & 16-23; lanes 16-31:
// K 8-15 & 24-31; B fragments read from W^T rows with the same pattern).
// WMMA sits in uniform control flow (EXEC all-ones requirement).
// Epilogue is UNCONDITIONAL: `out` must be padded to ceil(M/128)*128 rows
// (padding rows receive bias-only garbage and are never consumed).
// ---------------------------------------------------------------------------
template <int KP, bool RELU>
__global__ void __launch_bounds__(256)
k_gemm_gather_bf16(const __bf16* s0, const int* i0, int w0,
                   const __bf16* s1, const int* i1, int w1,
                   const __bf16* s2, const int* i2, int w2,
                   const __bf16* s3, const int* i3, int w3,
                   int M,
                   const __bf16* __restrict__ Wt, const float* __restrict__ bias,
                   int N,
                   __bf16* __restrict__ out) {
    constexpr int KSTEPS = KP / 32;
    extern __shared__ char smem[];
    __bf16* As = (__bf16*)smem;               // 128 * KP
    __bf16* Ws = As + 128 * KP;               // N  * KP

    const int tid = threadIdx.x;

    // ---- stage W^T into LDS (16B chunks) ----
    {
        const int nv = (N * KP) >> 3;
        const v8bf* src = (const v8bf*)Wt;
        v8bf* dst = (v8bf*)Ws;
        for (int i = tid; i < nv; i += 256) dst[i] = src[i];
    }

    // ---- stage A tile: 2 threads per row, each covers KP/2 columns ----
    {
        const int r_local = tid >> 1;
        const int kbeg = (tid & 1) * (KP >> 1);
        const int kend = kbeg + (KP >> 1);
        const int row  = blockIdx.x * 128 + r_local;
        __bf16* arow = As + (size_t)r_local * KP;

        int off = 0;
        if (row < M) {
            const __bf16* srcs[4] = { s0, s1, s2, s3 };
            const int*    idxs[4] = { i0, i1, i2, i3 };
            const int     wids[4] = { w0, w1, w2, w3 };
            for (int s = 0; s < 4; ++s) {
                int w = wids[s];
                if (!w) continue;
                int lo = kbeg > off ? kbeg : off;
                int hi = kend < off + w ? kend : off + w;
                if (lo < hi) {
                    int sr = idxs[s] ? idxs[s][row] : row;
                    const __bf16* sp = srcs[s] + (size_t)sr * w + (lo - off);
                    int cnt = hi - lo;
                    // all offsets/widths are multiples of 16 elems -> 16B aligned
                    v8bf* dv = (v8bf*)(arow + lo);
                    const v8bf* sv = (const v8bf*)sp;
                    int nv = cnt >> 3;
                    for (int k = 0; k < nv; ++k) dv[k] = sv[k];
                }
                off += w;
            }
        }
        // zero pad (rows >= M fully; valid rows from sum-of-widths to KP)
        int zlo = (row < M) ? (kbeg > off ? kbeg : off) : kbeg;
        if (zlo < kend) {
            v8bf zv = {};
            v8bf* dv = (v8bf*)(arow + zlo);
            int nv = (kend - zlo) >> 3;
            for (int k = 0; k < nv; ++k) dv[k] = zv;
        }
    }
    __syncthreads();

    // ---- WMMA compute ----
    const int wave = tid >> 5;
    const int lane = tid & 31;
    const int l16  = lane & 15;
    const int hiHalf = (lane >= 16);
    const int koff = hiHalf ? 8 : 0;

    // load A fragments once into registers (reused across all N tiles)
    v16bf afrag[KSTEPS];
    {
        const __bf16* arow = As + (size_t)(wave * 16 + l16) * KP;
#pragma unroll
        for (int s = 0; s < KSTEPS; ++s) {
            v8bf alo = *(const v8bf*)(arow + s * 32 + koff);
            v8bf ahi = *(const v8bf*)(arow + s * 32 + 16 + koff);
#pragma unroll
            for (int q = 0; q < 8; ++q) { afrag[s][q] = alo[q]; afrag[s][8 + q] = ahi[q]; }
        }
    }

    const int nt = N >> 4;
    for (int n = 0; n < nt; ++n) {
        const int col = n * 16 + l16;
        const __bf16* brow = Ws + (size_t)col * KP;
        v8f acc = {};
#pragma unroll
        for (int s = 0; s < KSTEPS; ++s) {
            v8bf blo = *(const v8bf*)(brow + s * 32 + koff);
            v8bf bhi = *(const v8bf*)(brow + s * 32 + 16 + koff);
            v16bf b;
#pragma unroll
            for (int q = 0; q < 8; ++q) { b[q] = blo[q]; b[8 + q] = bhi[q]; }
            acc = __builtin_amdgcn_wmma_f32_16x16x32_bf16(
                    /*neg_a=*/false, afrag[s], /*neg_b=*/false, b,
                    /*c_mod=*/(short)0, acc, /*reuse_a=*/false, /*reuse_b=*/false);
        }
        // branch-free epilogue: C layout — VGPR v: lanes0-15 M=v, lanes16-31 M=v+8
        const float bv = bias[col];
        const int rbase = blockIdx.x * 128 + wave * 16 + (hiHalf ? 8 : 0);
        __bf16* op = out + (size_t)rbase * N + col;
#pragma unroll
        for (int v = 0; v < 8; ++v) {
            float x = acc[v] + bv;
            if (RELU) x = fmaxf(x, 0.f);
            op[(size_t)v * N] = f2bf(x);
        }
    }
}

// ---------------------------------------------------------------------------
// Host orchestration
// ---------------------------------------------------------------------------
extern "C" void kernel_launch(void* const* d_in, const int* in_sizes, int n_in,
                              void* d_out, int out_size, void* d_ws, size_t ws_size,
                              hipStream_t stream) {
    (void)in_sizes; (void)n_in; (void)out_size; (void)ws_size;
    const int E = 800000, NN = 50000, NG = 250;
    const int NNp = (NN + 127) & ~127;   // 50048: node GEMM outputs padded
    const int NGp = (NG + 127) & ~127;   // 256:   global GEMM outputs padded
    // E is already a multiple of 128 (6250 * 128)

    const float* x_in  = (const float*)d_in[0];
    const int*   ei    = (const int*)d_in[1];
    const float* ea_in = (const float*)d_in[2];
    const float* u_in  = (const float*)d_in[3];
    const int*   batch = (const int*)d_in[4];
    const int* row = ei;
    const int* col = ei + E;

    // ---- workspace bump allocator ----
    char* base = (char*)d_ws;
    size_t cur = 0;
    auto alloc = [&](size_t bytes) -> void* {
        cur = (cur + 255) & ~(size_t)255;
        void* p = base + cur;
        cur += bytes;
        return p;
    };
    auto abf = [&](size_t elems) -> __bf16* { return (__bf16*)alloc(elems * 2); };
    auto af32 = [&](size_t elems) -> float* { return (float*)alloc(elems * 4); };

    // ---- pack all GEMM weights to bf16 W^T ----
    struct WDesc { int idx, K, N; };
    static const WDesc wd[] = {
        {5,16,128},{7,128,64},          // edge_first
        {9,32,128},{11,128,64},         // node_first
        {13,16,128},{15,128,64},        // global_first
        {17,256,128},{19,128,128},{21,128,64},   // blk0 edge
        {23,192,128},{25,128,128},{27,128,64},   // blk0 node
        {29,192,128},{31,128,128},{33,128,64},   // blk0 global
        {35,256,128},{37,128,128},{39,128,64},
        {41,192,128},{43,128,128},{45,128,64},
        {47,192,128},{49,128,128},{51,128,64},
        {53,256,128},{55,128,128},{57,128,64},
        {59,192,128},{61,128,128},{63,128,64},
        {65,192,128},{67,128,128},{69,128,64},
        {71,64,128},{73,128,64},{75,64,128},{77,128,64},   // edge_dense 0,1
        {79,64,128},{81,128,64},{83,64,128},{85,128,64},   // node_dense 0,1
        {87,64,128},{89,128,64},{91,64,128},{93,128,64},   // global_dense 0,1
    };
    __bf16* PW[96] = {};
    for (const WDesc& w : wd) {
        int Kp = (w.K + 31) & ~31;
        __bf16* dst = abf((size_t)w.N * Kp);
        PW[w.idx] = dst;
        int n = w.N * Kp;
        k_pack_wT<<<(n + 255) / 256, 256, 0, stream>>>(dst, (const float*)d_in[w.idx], w.K, w.N, Kp);
    }
    auto Bp = [&](int i) -> const float* { return (const float*)d_in[i]; };

    // ---- activation buffers (bf16 streams; GEMM outputs row-padded) ----
    __bf16* e_s   = abf((size_t)E * 64);
    __bf16* x_s   = abf((size_t)NNp * 64);
    __bf16* u_s   = abf((size_t)NGp * 64);
    __bf16* xin   = abf((size_t)NN * 32);
    __bf16* einb  = abf((size_t)E * 16);
    __bf16* uinb  = abf((size_t)NG * 16);
    int*    be    = (int*)alloc((size_t)E * 4);
    __bf16* h1    = abf((size_t)E * 128);
    __bf16* h2    = abf((size_t)E * 128);
    __bf16* eh2   = abf((size_t)E * 64);
    __bf16* ehin  = abf((size_t)E * 64);
    __bf16* nh1   = abf((size_t)NNp * 128);
    __bf16* nh2   = abf((size_t)NNp * 128);
    __bf16* xh2   = abf((size_t)NNp * 64);
    __bf16* xhinb = abf((size_t)NNp * 64);
    __bf16* gh1   = abf((size_t)NGp * 128);
    __bf16* gh2   = abf((size_t)NGp * 128);
    __bf16* uh2   = abf((size_t)NGp * 64);
    __bf16* uhinb = abf((size_t)NGp * 64);
    __bf16* aggbf = abf((size_t)NN * 64);
    __bf16* emean = abf((size_t)NG * 64);
    __bf16* xmean = abf((size_t)NG * 64);
    float*  aggN  = af32((size_t)NN * 64);
    float*  cntN  = af32((size_t)NN);
    float*  aggB  = af32((size_t)NG * 64);
    float*  cntB  = af32((size_t)NG);

    // ---- launch helpers ----
    struct Seg { const __bf16* s; const int* i; int w; };
    const Seg Z = { nullptr, nullptr, 0 };
    auto gemm = [&](Seg a, Seg b, Seg c, Seg d, int M, const __bf16* Wt,
                    const float* bias, int N, bool relu, __bf16* out) {
        int Ks = a.w + b.w + c.w + d.w;
        int Kp = (Ks + 31) & ~31;
        size_t lds = (size_t)(128 + N) * Kp * 2;
        dim3 g((M + 127) / 128);
#define GEMM_ARGS a.s,a.i,a.w, b.s,b.i,b.w, c.s,c.i,c.w, d.s,d.i,d.w, M, Wt, bias, N, out
#define GEMM_CASE(KPV) case KPV: \
        if (relu) k_gemm_gather_bf16<KPV, true ><<<g, 256, lds, stream>>>(GEMM_ARGS); \
        else      k_gemm_gather_bf16<KPV, false><<<g, 256, lds, stream>>>(GEMM_ARGS); \
        break;
        switch (Kp) {
        GEMM_CASE(32)
        GEMM_CASE(64)
        GEMM_CASE(128)
        GEMM_CASE(192)
        default:
        GEMM_CASE(256)
        }
#undef GEMM_CASE
#undef GEMM_ARGS
    };
    auto zero = [&](float* p, int n) { k_zero_f32<<<(n + 255) / 256, 256, 0, stream>>>(p, n); };

    // ---- input conversions ----
    k_f32_to_bf16<<<(NN * 32 + 255) / 256, 256, 0, stream>>>(xin, x_in, NN * 32);
    k_f32_to_bf16<<<(E * 16 + 255) / 256, 256, 0, stream>>>(einb, ea_in, E * 16);
    k_f32_to_bf16<<<(NG * 16 + 255) / 256, 256, 0, stream>>>(uinb, u_in, NG * 16);
    k_batch_e<<<(E + 255) / 256, 256, 0, stream>>>(be, row, batch, E);

    // ---- first MLPs ----
    gemm({einb, nullptr, 16}, Z, Z, Z, E, PW[5], Bp(6), 128, true, h1);
    gemm({h1, nullptr, 128}, Z, Z, Z, E, PW[7], Bp(8), 64, false, e_s);
    gemm({xin, nullptr, 32}, Z, Z, Z, NN, PW[9], Bp(10), 128, true, nh1);
    gemm({nh1, nullptr, 128}, Z, Z, Z, NN, PW[11], Bp(12), 64, false, x_s);
    gemm({uinb, nullptr, 16}, Z, Z, Z, NG, PW[13], Bp(14), 128, true, gh1);
    gemm({gh1, nullptr, 128}, Z, Z, Z, NG, PW[15], Bp(16), 64, false, u_s);

    // ---- message-passing blocks ----
    for (int b = 0; b < 3; ++b) {
        const __bf16 *eh, *xh, *uh;
        if (b == 0) {
            eh = e_s; xh = x_s; uh = u_s;
        } else {
            int ed = 71 + 4 * (b - 1), nd = 79 + 4 * (b - 1), gd = 87 + 4 * (b - 1);
            gemm({e_s, nullptr, 64}, Z, Z, Z, E, PW[ed], Bp(ed + 1), 128, true, h1);
            gemm({h1, nullptr, 128}, Z, Z, Z, E, PW[ed + 2], Bp(ed + 3), 64, false, ehin);
            gemm({x_s, nullptr, 64}, Z, Z, Z, NN, PW[nd], Bp(nd + 1), 128, true, nh1);
            gemm({nh1, nullptr, 128}, Z, Z, Z, NN, PW[nd + 2], Bp(nd + 3), 64, false, xhinb);
            gemm({u_s, nullptr, 64}, Z, Z, Z, NG, PW[gd], Bp(gd + 1), 128, true, gh1);
            gemm({gh1, nullptr, 128}, Z, Z, Z, NG, PW[gd + 2], Bp(gd + 3), 64, false, uhinb);
            eh = ehin; xh = xhinb; uh = uhinb;
        }
        const int eb = 17 + 18 * b, nb = eb + 6, gb = eb + 12;

        // edge model: cat(x[row], x[col], e, u[batch_e]) -> 128 -> 128 -> 64
        gemm({xh, row, 64}, {xh, col, 64}, {eh, nullptr, 64}, {uh, be, 64},
             E, PW[eb], Bp(eb + 1), 128, true, h1);
        gemm({h1, nullptr, 128}, Z, Z, Z, E, PW[eb + 2], Bp(eb + 3), 128, true, h2);
        gemm({h2, nullptr, 128}, Z, Z, Z, E, PW[eb + 4], Bp(eb + 5), 64, false, eh2);

        // node model: scatter_mean(eh2, row) onto nodes
        zero(aggN, NN * 64); zero(cntN, NN);
        k_scatter_feat<<<(E * 64 + 255) / 256, 256, 0, stream>>>(eh2, row, aggN, E);
        k_scatter_cnt<<<(E + 255) / 256, 256, 0, stream>>>(row, cntN, E);
        k_mean_bf16<<<(NN * 64 + 255) / 256, 256, 0, stream>>>(aggN, cntN, aggbf, NN);
        gemm({aggbf, nullptr, 64}, {xh, nullptr, 64}, {uh, batch, 64}, Z,
             NN, PW[nb], Bp(nb + 1), 128, true, nh1);
        gemm({nh1, nullptr, 128}, Z, Z, Z, NN, PW[nb + 2], Bp(nb + 3), 128, true, nh2);
        gemm({nh2, nullptr, 128}, Z, Z, Z, NN, PW[nb + 4], Bp(nb + 5), 64, false, xh2);

        // global model: per-graph means of eh2 and xh2
        zero(aggB, NG * 64); zero(cntB, NG);
        k_scatter_feat<<<(E * 64 + 255) / 256, 256, 0, stream>>>(eh2, be, aggB, E);
        k_scatter_cnt<<<(E + 255) / 256, 256, 0, stream>>>(be, cntB, E);
        k_mean_bf16<<<(NG * 64 + 255) / 256, 256, 0, stream>>>(aggB, cntB, emean, NG);
        zero(aggB, NG * 64); zero(cntB, NG);
        k_scatter_feat<<<(NN * 64 + 255) / 256, 256, 0, stream>>>(xh2, batch, aggB, NN);
        k_scatter_cnt<<<(NN + 255) / 256, 256, 0, stream>>>(batch, cntB, NN);
        k_mean_bf16<<<(NG * 64 + 255) / 256, 256, 0, stream>>>(aggB, cntB, xmean, NG);
        gemm({uh, nullptr, 64}, {xmean, nullptr, 64}, {emean, nullptr, 64}, Z,
             NG, PW[gb], Bp(gb + 1), 128, true, gh1);
        gemm({gh1, nullptr, 128}, Z, Z, Z, NG, PW[gb + 2], Bp(gb + 3), 128, true, gh2);
        gemm({gh2, nullptr, 128}, Z, Z, Z, NG, PW[gb + 4], Bp(gb + 5), 64, false, uh2);

        // outer residuals
        k_resadd<<<(NN * 64 + 255) / 256, 256, 0, stream>>>(x_s, xh2, NN * 64);
        k_resadd<<<(E * 64 + 255) / 256, 256, 0, stream>>>(e_s, eh2, E * 64);
        k_resadd<<<(NG * 64 + 255) / 256, 256, 0, stream>>>(u_s, uh2, NG * 64);
    }

    // ---- outputs: (xo, e, uo) concatenated, fp32 ----
    float* out = (float*)d_out;
    k_bf16_to_f32<<<(NN * 64 + 255) / 256, 256, 0, stream>>>(out, x_s, NN * 64);
    k_bf16_to_f32<<<(E * 64 + 255) / 256, 256, 0, stream>>>(out + (size_t)NN * 64, e_s, E * 64);
    k_bf16_to_f32<<<(NG * 64 + 255) / 256, 256, 0, stream>>>(out + (size_t)NN * 64 + (size_t)E * 64, u_s, NG * 64);
}
